// V_Layer_2121713844817
// MI455X (gfx1250) — matrix operand order repaired and checked by reading, not compiled, not run on hardware
//
#include <hip/hip_runtime.h>

typedef __attribute__((ext_vector_type(16))) _Float16 v16h;
typedef __attribute__((ext_vector_type(8)))  float    v8f;

// Problem constants (match reference): B=2, P=512, N=8, H=32
#define BB 2
#define PP 512
#define NN 8
#define HH 32
#define PT 4   // p-values per block (amortize W and e reads)

__global__ __launch_bounds__(256)
void v_layer_kernel(const float* __restrict__ d,
                    const float* __restrict__ e,
                    const float* __restrict__ W,
                    float* __restrict__ out) {
    // LDS: d vectors (broadcast) + tmp matrices transposed [p][k][j] in f16
    __shared__ float d_s[PT][HH];
    __shared__ __align__(32) _Float16 tmpT[PT][HH][HH];

    const int idx   = blockIdx.x;
    const int ptile = idx & (PP / PT - 1);          // 0..127
    const int n     = (idx >> 7) & (NN - 1);        // 0..7
    const int b     = idx >> 10;                    // 0..1
    const int p0    = ptile * PT;
    const int tid   = threadIdx.x;

    // ---- load d[b, p0..p0+3, n, :] into LDS --------------------------------
    if (tid < PT * HH) {
        const int pp = tid >> 5;
        const int i  = tid & 31;
        d_s[pp][i] = d[(((size_t)b * PP + (p0 + pp)) * NN + n) * HH + i];
    }
    __syncthreads();

    // ---- stage 1: tmpT[pp][k][j] = sum_i d[pp][i] * W[n,i,j,k] -------------
    {
        const int j  = tid >> 3;            // 0..31
        const int k4 = (tid & 7) * 4;       // 0,4,...,28
        const float* Wbase = W + (size_t)n * HH * HH * HH + (size_t)j * HH + k4;
        float acc[PT][4];
        #pragma unroll
        for (int pp = 0; pp < PT; ++pp)
            #pragma unroll
            for (int m = 0; m < 4; ++m) acc[pp][m] = 0.0f;
        #pragma unroll 4
        for (int i = 0; i < HH; ++i) {
            const float4 w4 = *(const float4*)(Wbase + (size_t)i * (HH * HH));
            #pragma unroll
            for (int pp = 0; pp < PT; ++pp) {
                const float dv = d_s[pp][i];
                acc[pp][0] += dv * w4.x;
                acc[pp][1] += dv * w4.y;
                acc[pp][2] += dv * w4.z;
                acc[pp][3] += dv * w4.w;
            }
        }
        #pragma unroll
        for (int pp = 0; pp < PT; ++pp)
            #pragma unroll
            for (int m = 0; m < 4; ++m)
                tmpT[pp][k4 + m][j] = (_Float16)acc[pp][m];
    }
    __syncthreads();

    // ---- stage 2: v[b,p,q,n,k] = sum_j e[b,q,n,j] * tmpT[p][k][j] ----------
    const int wave  = tid >> 5;     // 0..7
    const int lane  = tid & 31;
    const int lrow  = lane & 15;    // row (A) / column N (B,C)
    const int lhalf = lane >> 4;    // lane-group select

    #pragma unroll
    for (int t = 0; t < 4; ++t) {
        const int qbase = (wave + t * 8) * 16;      // 0..496
        const int q     = qbase + lrow;

        // A tile: 16x32 f16, layout: lanes 0-15 hold K=0..15,
        // lanes 16-31 hold K=16..31, row M = lane%16
        const float* erow = e + (((size_t)b * PP + q) * NN + n) * HH;
        const int jb = lhalf * 16;
        const float4 f0 = *(const float4*)(erow + jb);
        const float4 f1 = *(const float4*)(erow + jb + 4);
        const float4 f2 = *(const float4*)(erow + jb + 8);
        const float4 f3 = *(const float4*)(erow + jb + 12);
        v16h a;
        a[0]  = (_Float16)f0.x; a[1]  = (_Float16)f0.y;
        a[2]  = (_Float16)f0.z; a[3]  = (_Float16)f0.w;
        a[4]  = (_Float16)f1.x; a[5]  = (_Float16)f1.y;
        a[6]  = (_Float16)f1.z; a[7]  = (_Float16)f1.w;
        a[8]  = (_Float16)f2.x; a[9]  = (_Float16)f2.y;
        a[10] = (_Float16)f2.z; a[11] = (_Float16)f2.w;
        a[12] = (_Float16)f3.x; a[13] = (_Float16)f3.y;
        a[14] = (_Float16)f3.z; a[15] = (_Float16)f3.w;

        #pragma unroll
        for (int pp = 0; pp < PT; ++pp) {
            // B tiles: 32x16 f16, layout (same k order as A): column N = lane%16,
            // lanes 0-15 hold K=0..15, lanes 16-31 hold K=16..31 (contiguous j)
            const v16h b0 = *(const v16h*)&tmpT[pp][lrow][lhalf * 16];        // k-tile 0
            const v16h b1 = *(const v16h*)&tmpT[pp][16 + lrow][lhalf * 16];   // k-tile 1

            v8f c0 = {};
            v8f c1 = {};
            c0 = __builtin_amdgcn_wmma_f32_16x16x32_f16(
                false, a, false, b0, (short)0, c0, false, false);
            c1 = __builtin_amdgcn_wmma_f32_16x16x32_f16(
                false, a, false, b1, (short)0, c1, false, false);

            // C/D layout: VGPR r holds M = r + 8*(lane/16), N = lane%16
            const size_t pbase = ((size_t)b * PP + (p0 + pp)) * PP;
            #pragma unroll
            for (int r = 0; r < 8; ++r) {
                const int qq = qbase + lhalf * 8 + r;
                const size_t o = ((pbase + qq) * NN + n) * HH;
                __builtin_nontemporal_store(c0[r], out + o + lrow);        // k = lrow
                __builtin_nontemporal_store(c1[r], out + o + 16 + lrow);   // k = 16+lrow
            }
        }
    }
}

extern "C" void kernel_launch(void* const* d_in, const int* in_sizes, int n_in,
                              void* d_out, int out_size, void* d_ws, size_t ws_size,
                              hipStream_t stream) {
    (void)in_sizes; (void)n_in; (void)d_ws; (void)ws_size; (void)out_size;
    const float* d = (const float*)d_in[0];
    const float* e = (const float*)d_in[1];
    const float* W = (const float*)d_in[2];
    float* out = (float*)d_out;

    const int blocks = BB * NN * (PP / PT);   // 2*8*128 = 2048
    v_layer_kernel<<<blocks, 256, 0, stream>>>(d, e, W, out);
}
